// SWMSA_66864050864841
// MI455X (gfx1250) — matrix operand order repaired and checked by reading, not compiled, not run on hardware
//
#include <hip/hip_runtime.h>
#include <hip/hip_bf16.h>

// SW-MSA (Swin window attention), f16 WMMA pipeline for gfx1250 (wave32).
// GEMMs use GLOBAL_LOAD_ASYNC_TO_LDS_B128 (ASYNCcnt) with double-buffered LDS.

typedef __attribute__((ext_vector_type(16))) _Float16 v16h;
typedef __attribute__((ext_vector_type(8)))  _Float16 v8h;
typedef __attribute__((ext_vector_type(8)))  float    v8f;

static __device__ __forceinline__ v16h cat8(v8h lo, v8h hi) {
  return __builtin_shufflevector(lo, hi, 0,1,2,3,4,5,6,7,8,9,10,11,12,13,14,15);
}

// Async 16B copy global -> LDS, per lane. LDS generic pointers keep the raw
// LDS byte offset in the low 32 bits (aperture lives in addr[63:32]).
static __device__ __forceinline__ void async16(void* lds, const void* gptr) {
  asm volatile("global_load_async_to_lds_b128 %0, %1, off"
               :: "v"((unsigned)(uintptr_t)lds), "v"(gptr)
               : "memory");
}
static __device__ __forceinline__ void wait_async0() {
  asm volatile("s_wait_asynccnt 0x0" ::: "memory");
}

// ---------------------------------------------------------------------------
// Kernel 1: transpose weights to [N][K] f16 so WMMA B-fragments are contiguous.
__global__ __launch_bounds__(256) void prep_w(const float* __restrict__ wq,
                                              const float* __restrict__ wo,
                                              _Float16* __restrict__ wtq,
                                              _Float16* __restrict__ wto) {
  const int n = blockIdx.x;     // 0..1023
  const int k = threadIdx.x;    // 0..255
  if (n < 768) {
    wtq[(size_t)n * 256 + k] = (_Float16)wq[(size_t)k * 768 + n];
  } else {
    const int n2 = n - 768;
    wto[(size_t)n2 * 256 + k] = (_Float16)wo[(size_t)k * 256 + n2];
  }
}

// ---------------------------------------------------------------------------
// Kernel 2: cyclic shift + window partition: x[B,C,H,W] -> xw f16 [100352][256]
__global__ __launch_bounds__(256) void gather_win(const float* __restrict__ x,
                                                  _Float16* __restrict__ xw) {
  const int r = blockIdx.x;          // 0..100351  (window*49 + l)
  const int c = threadIdx.x;         // channel
  const int w = r / 49, l = r - 49 * w;
  const int b = w >> 6, wh = (w & 63) >> 3, ww = w & 7;
  const int mh = l / 7, mw = l - 7 * mh;
  int hx = wh * 7 + mh + 3; if (hx >= 56) hx -= 56;   // roll(-3)
  int wx = ww * 7 + mw + 3; if (wx >= 56) wx -= 56;
  xw[(size_t)r * 256 + c] =
      (_Float16)x[(((size_t)b * 256 + c) * 56 + hx) * 56 + wx];
}

// ---------------------------------------------------------------------------
// Kernel 3: QKV projection GEMM [100352,256]x[256,768] -> q/k/v f16 buffers.
// 256 threads / 8 waves; WG tile 128x128, wave tile 32x64, BK=32.
// Double-buffered LDS fed by async global->LDS copies.
__global__ __launch_bounds__(256) void qkv_gemm(const _Float16* __restrict__ xw,
                                                const _Float16* __restrict__ wtq,
                                                const float* __restrict__ bqkv,
                                                _Float16* __restrict__ qb,
                                                _Float16* __restrict__ kbuf,
                                                _Float16* __restrict__ vb) {
  __shared__ __align__(32) _Float16 As[2][128 * 32];
  __shared__ __align__(32) _Float16 Bs[2][128 * 32];
  const int r0 = blockIdx.x * 128;
  const int n0 = blockIdx.y * 128;
  const int t = threadIdx.x;
  const int lane = t & 31, wv = t >> 5;
  const int l15 = lane & 15, g = lane >> 4;
  const int wm = (wv & 3) * 32;   // wave m-offset in WG tile
  const int wn = (wv >> 2) * 64;  // wave n-offset in WG tile

  auto fill = [&](int buf, int kc) {
#pragma unroll
    for (int it = 0; it < 2; ++it) {
      const int idx = t + 256 * it;        // 0..511 : 128 rows x 4 segs(16B)
      const int row = idx >> 2, part = idx & 3;
      async16(&As[buf][row * 32 + part * 8],
              xw + (size_t)(r0 + row) * 256 + kc + part * 8);
      async16(&Bs[buf][row * 32 + part * 8],
              wtq + (size_t)(n0 + row) * 256 + kc + part * 8);
    }
  };

  v8f acc[2][4] = {};
  fill(0, 0);
  for (int i = 0; i < 8; ++i) {
    const int cur = i & 1;
    wait_async0();          // this wave's async fills done
    __syncthreads();        // all waves' fills done; prev compute done
    if (i < 7) fill(1 - cur, (i + 1) * 32);
    v16h a[2];
#pragma unroll
    for (int mt = 0; mt < 2; ++mt) {
      const _Float16* ap = &As[cur][(wm + 16 * mt + l15) * 32 + g * 8];
      a[mt] = cat8(*(const v8h*)ap, *(const v8h*)(ap + 16));
    }
#pragma unroll
    for (int nt = 0; nt < 4; ++nt) {
      const _Float16* bp = &Bs[cur][(wn + 16 * nt + l15) * 32 + g * 8];
      const v16h b = cat8(*(const v8h*)bp, *(const v8h*)(bp + 16));
#pragma unroll
      for (int mt = 0; mt < 2; ++mt) {
        acc[mt][nt] = __builtin_amdgcn_wmma_f32_16x16x32_f16(
            false, a[mt], false, b, (short)0, acc[mt][nt], false, false);
      }
    }
  }

  const float qscale = 0.17677669529663687f;  // 1/sqrt(32)
#pragma unroll
  for (int nt = 0; nt < 4; ++nt) {
    const int n = n0 + wn + 16 * nt + l15;
    const int which = n >> 8;                 // uniform within 16-col tile
    const int c = n & 255;
    const int h = c >> 5, d = c & 31;
    const float bias = bqkv[n];
    _Float16* dst = (which == 0) ? qb : (which == 1 ? kbuf : vb);
#pragma unroll
    for (int mt = 0; mt < 2; ++mt) {
#pragma unroll
      for (int vg = 0; vg < 8; ++vg) {
        const int r = r0 + wm + 16 * mt + vg + 8 * g;
        const int w = r / 49, l = r - 49 * w;
        float val = acc[mt][nt][vg] + bias;
        if (which == 0) val *= qscale;
        dst[((size_t)(w * 8 + h) * 49 + l) * 32 + d] = (_Float16)val;
      }
    }
  }
}

// ---------------------------------------------------------------------------
// Kernel 4: attention per (window, head). L=49 padded to 64, d=32.
__global__ __launch_bounds__(128) void attn(const _Float16* __restrict__ qb,
                                            const _Float16* __restrict__ kbuf,
                                            const _Float16* __restrict__ vb,
                                            _Float16* __restrict__ ob) {
  __shared__ __align__(32) _Float16 Qs[64 * 32];
  __shared__ __align__(32) _Float16 Ks[64 * 32];
  __shared__ __align__(32) _Float16 Vt[32 * 64];
  __shared__ __align__(32) _Float16 Ps[64 * 64];
  const int w = blockIdx.x, h = blockIdx.y;
  const size_t base = (size_t)(w * 8 + h) * 49 * 32;
  const int t = threadIdx.x, lane = t & 31, wv = t >> 5;
  const int l15 = lane & 15, g = lane >> 4;

  // Issue all global loads first (no interleaved waits), then LDS stores.
  v8h qreg[2], kreg[2], vreg[2];
#pragma unroll
  for (int it = 0; it < 2; ++it) {
    const int s = t + 128 * it;              // 0..255 : 64 rows x 4 segs
    const int row = s >> 2, part = s & 3;
    const v8h z = {0, 0, 0, 0, 0, 0, 0, 0};
    qreg[it] = z; kreg[it] = z; vreg[it] = z;
    if (row < 49) {
      qreg[it] = *(const v8h*)(qb   + base + row * 32 + part * 8);
      kreg[it] = *(const v8h*)(kbuf + base + row * 32 + part * 8);
      vreg[it] = *(const v8h*)(vb   + base + row * 32 + part * 8);
    }
  }
#pragma unroll
  for (int it = 0; it < 2; ++it) {
    const int s = t + 128 * it;
    const int row = s >> 2, part = s & 3;
    *(v8h*)(Qs + row * 32 + part * 8) = qreg[it];
    *(v8h*)(Ks + row * 32 + part * 8) = kreg[it];
#pragma unroll
    for (int e = 0; e < 8; ++e)              // transpose V into Vt[d][l]
      Vt[(part * 8 + e) * 64 + row] = vreg[it][e];
  }
  __syncthreads();

  // S = Q @ K^T  (this wave: rows 16*wv..16*wv+15, all 64 cols; k-dim = 32)
  const v8h qlo = *(const v8h*)(Qs + (16 * wv + l15) * 32 + g * 8);
  const v8h qhi = *(const v8h*)(Qs + (16 * wv + l15) * 32 + g * 8 + 16);
  const v16h qa = cat8(qlo, qhi);
  v8f s[4] = {};
#pragma unroll
  for (int j = 0; j < 4; ++j) {
    const v16h kb = cat8(*(const v8h*)(Ks + (16 * j + l15) * 32 + g * 8),
                         *(const v8h*)(Ks + (16 * j + l15) * 32 + g * 8 + 16));
    s[j] = __builtin_amdgcn_wmma_f32_16x16x32_f16(
        false, qa, false, kb, (short)0, s[j], false, false);
  }

  // Row-wise masked softmax. Row m lives in one VGPR across one 16-lane half.
  float rsum[8];
#pragma unroll
  for (int vg = 0; vg < 8; ++vg) {
    float m = -1e30f;
#pragma unroll
    for (int j = 0; j < 4; ++j) {
      const int col = 16 * j + l15;
      float val = s[j][vg];
      if (col >= 49) { val = -1e30f; s[j][vg] = val; }
      m = fmaxf(m, val);
    }
    m = fmaxf(m, __shfl_xor(m, 1));
    m = fmaxf(m, __shfl_xor(m, 2));
    m = fmaxf(m, __shfl_xor(m, 4));
    m = fmaxf(m, __shfl_xor(m, 8));
    float sum = 0.0f;
    const int row = 16 * wv + vg + 8 * g;
#pragma unroll
    for (int j = 0; j < 4; ++j) {
      const float p = __expf(s[j][vg] - m);
      sum += p;
      Ps[row * 64 + 16 * j + l15] = (_Float16)p;
    }
    sum += __shfl_xor(sum, 1);
    sum += __shfl_xor(sum, 2);
    sum += __shfl_xor(sum, 4);
    sum += __shfl_xor(sum, 8);
    rsum[vg] = sum;
  }
  __syncthreads();

  // O = P @ V  (k-dim 64 -> two 32-steps; d = 32 -> two n-tiles)
  v8f o[2] = {};
#pragma unroll
  for (int s2 = 0; s2 < 2; ++s2) {
    const v8h plo = *(const v8h*)(Ps + (16 * wv + l15) * 64 + 32 * s2 + g * 8);
    const v8h phi = *(const v8h*)(Ps + (16 * wv + l15) * 64 + 32 * s2 + g * 8 + 16);
    const v16h pa = cat8(plo, phi);
#pragma unroll
    for (int j = 0; j < 2; ++j) {
      const v16h bv = cat8(*(const v8h*)(Vt + (16 * j + l15) * 64 + 32 * s2 + g * 8),
                           *(const v8h*)(Vt + (16 * j + l15) * 64 + 32 * s2 + g * 8 + 16));
      o[j] = __builtin_amdgcn_wmma_f32_16x16x32_f16(
          false, pa, false, bv, (short)0, o[j], false, false);
    }
  }

  // Store O rows < 49, divided by softmax denom: ob[win][l][h*32+d] f16
#pragma unroll
  for (int j = 0; j < 2; ++j) {
#pragma unroll
    for (int vg = 0; vg < 8; ++vg) {
      const int l = 16 * wv + vg + 8 * g;
      if (l < 49) {
        const float val = o[j][vg] / rsum[vg];
        ob[((size_t)w * 49 + l) * 256 + h * 32 + 16 * j + l15] = (_Float16)val;
      }
    }
  }
}

// ---------------------------------------------------------------------------
// Kernel 5: output projection [100352,256]x[256,256] + bias, scatter to
// [B,C,H,W] fp32 with window merge and reverse cyclic shift.
__global__ __launch_bounds__(256) void out_gemm(const _Float16* __restrict__ ab,
                                                const _Float16* __restrict__ wto,
                                                const float* __restrict__ bo,
                                                float* __restrict__ out) {
  __shared__ __align__(32) _Float16 As[2][128 * 32];
  __shared__ __align__(32) _Float16 Bs[2][128 * 32];
  const int r0 = blockIdx.x * 128;
  const int n0 = blockIdx.y * 128;
  const int t = threadIdx.x;
  const int lane = t & 31, wv = t >> 5;
  const int l15 = lane & 15, g = lane >> 4;
  const int wm = (wv & 3) * 32;
  const int wn = (wv >> 2) * 64;

  auto fill = [&](int buf, int kc) {
#pragma unroll
    for (int it = 0; it < 2; ++it) {
      const int idx = t + 256 * it;
      const int row = idx >> 2, part = idx & 3;
      async16(&As[buf][row * 32 + part * 8],
              ab + (size_t)(r0 + row) * 256 + kc + part * 8);
      async16(&Bs[buf][row * 32 + part * 8],
              wto + (size_t)(n0 + row) * 256 + kc + part * 8);
    }
  };

  v8f acc[2][4] = {};
  fill(0, 0);
  for (int i = 0; i < 8; ++i) {
    const int cur = i & 1;
    wait_async0();
    __syncthreads();
    if (i < 7) fill(1 - cur, (i + 1) * 32);
    v16h a[2];
#pragma unroll
    for (int mt = 0; mt < 2; ++mt) {
      const _Float16* ap = &As[cur][(wm + 16 * mt + l15) * 32 + g * 8];
      a[mt] = cat8(*(const v8h*)ap, *(const v8h*)(ap + 16));
    }
#pragma unroll
    for (int nt = 0; nt < 4; ++nt) {
      const _Float16* bp = &Bs[cur][(wn + 16 * nt + l15) * 32 + g * 8];
      const v16h b = cat8(*(const v8h*)bp, *(const v8h*)(bp + 16));
#pragma unroll
      for (int mt = 0; mt < 2; ++mt) {
        acc[mt][nt] = __builtin_amdgcn_wmma_f32_16x16x32_f16(
            false, a[mt], false, b, (short)0, acc[mt][nt], false, false);
      }
    }
  }

#pragma unroll
  for (int nt = 0; nt < 4; ++nt) {
    const int n = n0 + wn + 16 * nt + l15;   // output channel 0..255
    const float bias = bo[n];
#pragma unroll
    for (int mt = 0; mt < 2; ++mt) {
#pragma unroll
      for (int vg = 0; vg < 8; ++vg) {
        const int r = r0 + wm + 16 * mt + vg + 8 * g;
        const int w = r / 49, l = r - 49 * w;
        const int b = w >> 6, wh = (w & 63) >> 3, ww = w & 7;
        const int mh = l / 7, mw = l - 7 * mh;
        int hx = wh * 7 + mh + 3; if (hx >= 56) hx -= 56;  // reverse shift
        int wx = ww * 7 + mw + 3; if (wx >= 56) wx -= 56;
        out[(((size_t)b * 256 + n) * 56 + hx) * 56 + wx] = acc[mt][nt][vg] + bias;
      }
    }
  }
}

// ---------------------------------------------------------------------------
extern "C" void kernel_launch(void* const* d_in, const int* in_sizes, int n_in,
                              void* d_out, int out_size, void* d_ws, size_t ws_size,
                              hipStream_t stream) {
  (void)in_sizes; (void)n_in; (void)out_size; (void)ws_size;
  const float* x    = (const float*)d_in[0];   // [32,256,56,56]
  const float* wqkv = (const float*)d_in[1];   // [256,768]
  const float* bqkv = (const float*)d_in[2];   // [768]
  const float* wout = (const float*)d_in[3];   // [256,256]
  const float* bout = (const float*)d_in[4];   // [256]
  float* out = (float*)d_out;

  // Workspace layout (bytes):
  //   wtq  : 768*256*2            =    393,216
  //   wto  : 256*256*2            =    131,072
  //   xw   : 100352*256*2         = 51,380,224
  //   q,k,v: 3 * 2048*8*49*32*2   = 3*51,380,224
  //   ob   : 100352*256*2         = 51,380,224       total ~257.4 MB
  char* ws = (char*)d_ws;
  _Float16* wtq = (_Float16*)(ws);
  _Float16* wto = (_Float16*)(ws + 393216);
  _Float16* xw  = (_Float16*)(ws + 524288);
  _Float16* qb  = (_Float16*)(ws + 524288 + 51380224);
  _Float16* kb  = qb + 25690112;
  _Float16* vb  = kb + 25690112;
  _Float16* ob  = vb + 25690112;

  prep_w<<<dim3(1024), dim3(256), 0, stream>>>(wqkv, wout, wtq, wto);
  gather_win<<<dim3(100352), dim3(256), 0, stream>>>(x, xw);
  qkv_gemm<<<dim3(784, 6), dim3(256), 0, stream>>>(xw, wtq, bqkv, qb, kb, vb);
  attn<<<dim3(2048, 8), dim3(128), 0, stream>>>(qb, kb, vb, ob);
  out_gemm<<<dim3(784, 2), dim3(256), 0, stream>>>(ob, wto, bout, out);
}
